// DeepSigNet_31945966747673
// MI455X (gfx1250) — compile-verified, matrix operand order checked by reading, not run on hardware
//
#include <hip/hip_runtime.h>

typedef unsigned short u16;
typedef unsigned int   u32;

typedef __attribute__((ext_vector_type(16))) __bf16 v16bf;
typedef __attribute__((ext_vector_type(8)))  float  v8f;

#define BB 16
#define LL 128
#define HH 20
#define GG 80            // 4*H gates
#define MM (BB * LL)     // 2048 rows
#define K1P 25280        // 25259 padded to /32
#define K2P 8448         // 8420  padded to /32
#define S1 25259
#define S2 8420
#define NSPLIT 16        // K-split factor for the WMMA GEMM

__device__ __forceinline__ u16 f2bf(float f) {
  u32 u = __float_as_uint(f);
  u32 r = u + 0x7FFFu + ((u >> 16) & 1u);   // round-to-nearest-even
  return (u16)(r >> 16);
}
__device__ __forceinline__ float bf2f(u16 h) {
  return __uint_as_float(((u32)h) << 16);
}
__device__ __forceinline__ float sigm(float x) { return 1.0f / (1.0f + expf(-x)); }

// ---------------- augment: hid = relu(x@w1^T+b1)@w2^T+b2 ; a = [t, x, hid] ----
__global__ void augment_k(const float* __restrict__ x, const float* __restrict__ w1,
                          const float* __restrict__ b1, const float* __restrict__ w2,
                          const float* __restrict__ b2, float* __restrict__ aug) {
  int pos = blockIdx.x * 256 + threadIdx.x;
  if (pos >= MM) return;
  float xin[20];
#pragma unroll
  for (int c = 0; c < 20; ++c) xin[c] = x[(size_t)pos * 20 + c];
  float hid[8];
#pragma unroll
  for (int o = 0; o < 8; ++o) hid[o] = b2[o];
  for (int h = 0; h < 64; ++h) {
    float a = b1[h];
#pragma unroll
    for (int c = 0; c < 20; ++c) a += xin[c] * w1[h * 20 + c];
    a = fmaxf(a, 0.0f);
#pragma unroll
    for (int o = 0; o < 8; ++o) hid[o] += a * w2[o * 64 + h];
  }
  float* dst = aug + (size_t)pos * 32;
  dst[0] = (float)(pos & (LL - 1)) * (1.0f / (LL - 1));
#pragma unroll
  for (int c = 0; c < 20; ++c) dst[1 + c] = xin[c];
#pragma unroll
  for (int o = 0; o < 8; ++o) dst[21 + o] = hid[o];
  dst[29] = 0.0f; dst[30] = 0.0f; dst[31] = 0.0f;
}

// ---------------- streamed depth-3 signature (Chen), one block per sequence ---
// s1,s2 in LDS; s3 elements thread-private; output bf16 rows of width Kp.
template <int C>
__global__ __launch_bounds__(512) void sig_scan(const float* __restrict__ x, int xstride,
                                                u16* __restrict__ out, int Kp) {
  constexpr int C2 = C * C, C3 = C * C * C;
  constexpr int NPT = (C3 + 511) / 512;
  __shared__ float s1[C], s2[C2], dsh[C];
  const int b = blockIdx.x, tid = threadIdx.x;
  float s3r[NPT];
#pragma unroll
  for (int e = 0; e < NPT; ++e) s3r[e] = 0.0f;
  for (int cc = tid; cc < C2; cc += 512) s2[cc] = 0.0f;
  if (tid < C) s1[tid] = 0.0f;
  __syncthreads();
  for (int l = 0; l < LL; ++l) {
    const int pos = b * LL + l;
    if (tid < C) {
      float cur = x[(size_t)pos * xstride + tid];
      float prev = (l == 0) ? 0.0f : x[(size_t)(pos - 1) * xstride + tid];
      dsh[tid] = cur - prev;
    }
    for (int p = C + C2 + C3 + tid; p < Kp; p += 512)   // K padding
      out[(size_t)pos * Kp + p] = 0;
    __syncthreads();
    // depth-3 (reads OLD s1, s2)
#pragma unroll
    for (int e = 0; e < NPT; ++e) {
      int idx = tid + e * 512;
      if (idx < C3) {
        int i = idx / C2, rem = idx % C2;
        int j = rem / C, k = rem % C;
        float dj = dsh[j], dk = dsh[k];
        float v = s3r[e] + dsh[i] * dj * dk * (1.0f / 6.0f)
                + s1[i] * 0.5f * dj * dk + s2[i * C + j] * dk;
        s3r[e] = v;
        out[(size_t)pos * Kp + C + C2 + idx] = f2bf(v);
      }
    }
    __syncthreads();
    // depth-2 (reads OLD s1)
    for (int cc = tid; cc < C2; cc += 512) {
      int i = cc / C, j = cc % C;
      float v = s2[cc] + 0.5f * dsh[i] * dsh[j] + s1[i] * dsh[j];
      s2[cc] = v;
      out[(size_t)pos * Kp + C + cc] = f2bf(v);
    }
    __syncthreads();
    if (tid < C) {
      float v = s1[tid] + dsh[tid];
      s1[tid] = v;
      out[(size_t)pos * Kp + tid] = f2bf(v);
    }
    __syncthreads();
  }
}

// ---------------- per-channel batch stats over the bf16 stream ----------------
__global__ void col_stats(const u16* __restrict__ X, int Kp,
                          float* __restrict__ mean, float* __restrict__ inv) {
  int c = blockIdx.x * 256 + threadIdx.x;
  if (c >= Kp) return;
  float s = 0.0f, ss = 0.0f;
  for (int r = 0; r < MM; ++r) {
    float v = bf2f(X[(size_t)r * Kp + c]);
    s += v; ss += v * v;
  }
  float m = s * (1.0f / MM);
  float var = ss * (1.0f / MM) - m * m;
  mean[c] = m;
  inv[c] = rsqrtf(fmaxf(var, 0.0f) + 1e-5f);
}

// ---------------- fold BN scale into the gate weights (bf16) ------------------
__global__ void fold_w(const float* __restrict__ wih, const float* __restrict__ inv,
                       int S, int Kp, u16* __restrict__ wf) {
  int k = blockIdx.x * 256 + threadIdx.x;
  int g = blockIdx.y;
  if (k >= Kp) return;
  wf[(size_t)g * Kp + k] = (k < S) ? f2bf(wih[(size_t)g * S + k] * inv[k]) : (u16)0;
}

// biasf[g] = bih+bhh - sum_k mean*inv*W
__global__ void fold_bias(const float* __restrict__ wih, const float* __restrict__ mean,
                          const float* __restrict__ inv, const float* __restrict__ bih,
                          const float* __restrict__ bhh, int S, float* __restrict__ biasf) {
  __shared__ float red[256];
  int g = blockIdx.x, tid = threadIdx.x;
  float s = 0.0f;
  for (int k = tid; k < S; k += 256) s += mean[k] * inv[k] * wih[(size_t)g * S + k];
  red[tid] = s;
  __syncthreads();
  for (int off = 128; off > 0; off >>= 1) {
    if (tid < off) red[tid] += red[tid + off];
    __syncthreads();
  }
  if (tid == 0) biasf[g] = bih[g] + bhh[g] - red[0];
}

// -------- split-K WMMA GEMM: part[sp](M,80) = A(M,k0:k1)bf16 x Wf(80,k0:k1)^T -
// grid = (M/16, NSPLIT), one wave32 per (M-tile, K-chunk): 2048 concurrent waves
// streaming the bf16 activation tensor -> enough MLP to approach HBM bandwidth.
union FragBf { uint4 u[2]; v16bf v; };

__device__ __forceinline__ void wmma_kstep(const u16* __restrict__ arow,
                                           const u16* __restrict__ Wf,
                                           int lo, int half, int k0, int Kp,
                                           v8f acc[5]) {
  FragBf a;
  a.u[0] = *(const uint4*)(arow + k0);
  a.u[1] = *(const uint4*)(arow + k0 + 16);
  __builtin_prefetch(arow + k0 + 256, 0, 1);   // global_prefetch_b8
#pragma unroll
  for (int t = 0; t < 5; ++t) {                // 5 N-tiles = 80 gates, reuse A
    const u16* brow = Wf + (size_t)(t * 16 + lo) * Kp + half * 8 + k0;
    FragBf b;
    b.u[0] = *(const uint4*)(brow);
    b.u[1] = *(const uint4*)(brow + 16);
    acc[t] = __builtin_amdgcn_wmma_f32_16x16x32_bf16(
        false, a.v, false, b.v, (short)0, acc[t], false, false);
  }
}

__global__ __launch_bounds__(32) void gemm_wmma_split(const u16* __restrict__ A,
                                                      const u16* __restrict__ Wf,
                                                      float* __restrict__ part, int Kp) {
  const int m0 = blockIdx.x * 16;
  const int sp = blockIdx.y;
  const int steps = Kp / 32;
  const int chunk = (steps + NSPLIT - 1) / NSPLIT;
  const int kb = sp * chunk * 32;
  const int ke = min(Kp, kb + chunk * 32);

  const int ln = threadIdx.x;
  const int half = ln >> 4, lo = ln & 15;
  // 16-bit A layout: lanes 0-15 -> M=lane, K {0..7,16..23}; lanes 16-31 -> K {8..15,24..31}
  const u16* arow = A + (size_t)(m0 + lo) * Kp + half * 8;
  v8f acc[5] = {};
  int k0 = kb;
  // unroll x2 so the 2nd step's A/B loads are in flight under the 1st step's WMMAs
  for (; k0 + 64 <= ke; k0 += 64) {
    wmma_kstep(arow, Wf, lo, half, k0, Kp, acc);
    wmma_kstep(arow, Wf, lo, half, k0 + 32, Kp, acc);
  }
  for (; k0 < ke; k0 += 32)
    wmma_kstep(arow, Wf, lo, half, k0, Kp, acc);

  // C layout: lane(0-15)->N, VGPR r -> M = r (+8 for lanes 16-31)
  const int rbase = m0 + (half ? 8 : 0);
  float* dst = part + (size_t)sp * MM * GG;
#pragma unroll
  for (int t = 0; t < 5; ++t)
#pragma unroll
    for (int r = 0; r < 8; ++r) {
      int row = rbase + r, col = t * 16 + lo;
      dst[(size_t)row * GG + col] = acc[t][r];
    }
}

// deterministic reduction of K-split partials + folded bias
__global__ void reduce_gates(const float* __restrict__ part, const float* __restrict__ biasf,
                             float* __restrict__ gates) {
  int idx = blockIdx.x * 256 + threadIdx.x;   // over MM*GG
  if (idx >= MM * GG) return;
  float s = biasf[idx % GG];
#pragma unroll
  for (int sp = 0; sp < NSPLIT; ++sp) s += part[(size_t)sp * MM * GG + idx];
  gates[idx] = s;
}

// ---------------- fused 2-layer LSTM scan (one wave per batch sequence) -------
__global__ __launch_bounds__(512) void lstm2_scan(
    const float* __restrict__ gates0, const float* __restrict__ whh0,
    const float* __restrict__ wih1, const float* __restrict__ whh1,
    const float* __restrict__ bih1, const float* __restrict__ bhh1,
    float* __restrict__ outh) {
  __shared__ float Wh0[GG * HH], Wi1[GG * HH], Wh1[GG * HH], B1[GG];
  __shared__ float h0[BB][HH], c0[BB][HH], h1[BB][HH], c1[BB][HH];
  __shared__ float gb[BB][GG];
  int tid = threadIdx.x;
  for (int i = tid; i < GG * HH; i += 512) { Wh0[i] = whh0[i]; Wi1[i] = wih1[i]; Wh1[i] = whh1[i]; }
  if (tid < GG) B1[tid] = bih1[tid] + bhh1[tid];
  int wv = tid >> 5, ln = tid & 31;
  if (ln < HH) { h0[wv][ln] = 0; c0[wv][ln] = 0; h1[wv][ln] = 0; c1[wv][ln] = 0; }
  __syncthreads();
  for (int l = 0; l < LL; ++l) {
    int pos = wv * LL + l;
    for (int g = ln; g < GG; g += 32) {          // layer-0 gates
      float acc = gates0[(size_t)pos * GG + g];
#pragma unroll
      for (int c = 0; c < HH; ++c) acc += h0[wv][c] * Wh0[g * HH + c];
      gb[wv][g] = acc;
    }
    __syncthreads();
    if (ln < HH) {
      float i_ = sigm(gb[wv][ln]);
      float f_ = sigm(gb[wv][HH + ln]);
      float g_ = tanhf(gb[wv][2 * HH + ln]);
      float o_ = sigm(gb[wv][3 * HH + ln]);
      float c = f_ * c0[wv][ln] + i_ * g_;
      c0[wv][ln] = c;
      h0[wv][ln] = o_ * tanhf(c);
    }
    __syncthreads();
    for (int g = ln; g < GG; g += 32) {          // layer-1 gates
      float acc = B1[g];
#pragma unroll
      for (int c = 0; c < HH; ++c) acc += h0[wv][c] * Wi1[g * HH + c] + h1[wv][c] * Wh1[g * HH + c];
      gb[wv][g] = acc;
    }
    __syncthreads();
    if (ln < HH) {
      float i_ = sigm(gb[wv][ln]);
      float f_ = sigm(gb[wv][HH + ln]);
      float g_ = tanhf(gb[wv][2 * HH + ln]);
      float o_ = sigm(gb[wv][3 * HH + ln]);
      float c = f_ * c1[wv][ln] + i_ * g_;
      c1[wv][ln] = c;
      float h = o_ * tanhf(c);
      h1[wv][ln] = h;
      outh[(size_t)pos * HH + ln] = h;
    }
    __syncthreads();
  }
}

// ---------------- head: LeakyReLU(0.01) + linear -> (B,L,1) -------------------
__global__ void head_k(const float* __restrict__ o3, const float* __restrict__ lw,
                       const float* __restrict__ lb, float* __restrict__ out) {
  int pos = blockIdx.x * 256 + threadIdx.x;
  if (pos >= MM) return;
  float s = lb[0];
#pragma unroll
  for (int c = 0; c < HH; ++c) {
    float v = o3[(size_t)pos * HH + c];
    v = (v >= 0.0f) ? v : 0.01f * v;
    s += v * lw[c];
  }
  out[pos] = s;
}

extern "C" void kernel_launch(void* const* d_in, const int* in_sizes, int n_in,
                              void* d_out, int out_size, void* d_ws, size_t ws_size,
                              hipStream_t stream) {
  (void)in_sizes; (void)n_in; (void)out_size; (void)ws_size;
  const float* inp = (const float*)d_in[0];
  const float* aw1 = (const float*)d_in[1];
  const float* ab1 = (const float*)d_in[2];
  const float* aw2 = (const float*)d_in[3];
  const float* ab2 = (const float*)d_in[4];
  const float* P[3][8];   // per stage: wih0,whh0,bih0,bhh0,wih1,whh1,bih1,bhh1
  for (int s = 0; s < 3; ++s)
    for (int i = 0; i < 8; ++i)
      P[s][i] = (const float*)d_in[5 + s * 8 + i];
  const float* lin_w = (const float*)d_in[29];
  const float* lin_b = (const float*)d_in[30];

  char* wsp = (char*)d_ws;
  auto take = [&](size_t bytes) { char* p = wsp; wsp += (bytes + 255) & ~(size_t)255; return p; };
  u16*   sig   = (u16*)take((size_t)MM * K1P * 2);           // bf16 signature stream (aliased per stage)
  u16*   wf    = (u16*)take((size_t)GG * K1P * 2);           // folded bf16 weights
  float* part  = (float*)take((size_t)NSPLIT * MM * GG * 4); // split-K partials
  float* mean  = (float*)take((size_t)K1P * 4);
  float* invs  = (float*)take((size_t)K1P * 4);
  float* biasf = (float*)take((size_t)GG * 4);
  float* aug   = (float*)take((size_t)MM * 32 * 4);
  float* gates = (float*)take((size_t)MM * GG * 4);
  float* o1    = (float*)take((size_t)MM * HH * 4);
  float* o2    = (float*)take((size_t)MM * HH * 4);
  float* o3    = (float*)take((size_t)MM * HH * 4);

  augment_k<<<MM / 256, 256, 0, stream>>>(inp, aw1, ab1, aw2, ab2, aug);

  // ---- stage 1: sig(29ch) -> BN-folded split-K WMMA GEMM -> LSTM2 ----
  sig_scan<29><<<BB, 512, 0, stream>>>(aug, 32, sig, K1P);
  col_stats<<<(K1P + 255) / 256, 256, 0, stream>>>(sig, K1P, mean, invs);
  fold_w<<<dim3((K1P + 255) / 256, GG), 256, 0, stream>>>(P[0][0], invs, S1, K1P, wf);
  fold_bias<<<GG, 256, 0, stream>>>(P[0][0], mean, invs, P[0][2], P[0][3], S1, biasf);
  gemm_wmma_split<<<dim3(MM / 16, NSPLIT), 32, 0, stream>>>(sig, wf, part, K1P);
  reduce_gates<<<(MM * GG + 255) / 256, 256, 0, stream>>>(part, biasf, gates);
  lstm2_scan<<<1, 512, 0, stream>>>(gates, P[0][1], P[0][4], P[0][5], P[0][6], P[0][7], o1);

  // ---- stage 2 ----
  sig_scan<20><<<BB, 512, 0, stream>>>(o1, 20, sig, K2P);
  col_stats<<<(K2P + 255) / 256, 256, 0, stream>>>(sig, K2P, mean, invs);
  fold_w<<<dim3((K2P + 255) / 256, GG), 256, 0, stream>>>(P[1][0], invs, S2, K2P, wf);
  fold_bias<<<GG, 256, 0, stream>>>(P[1][0], mean, invs, P[1][2], P[1][3], S2, biasf);
  gemm_wmma_split<<<dim3(MM / 16, NSPLIT), 32, 0, stream>>>(sig, wf, part, K2P);
  reduce_gates<<<(MM * GG + 255) / 256, 256, 0, stream>>>(part, biasf, gates);
  lstm2_scan<<<1, 512, 0, stream>>>(gates, P[1][1], P[1][4], P[1][5], P[1][6], P[1][7], o2);

  // ---- stage 3 ----
  sig_scan<20><<<BB, 512, 0, stream>>>(o2, 20, sig, K2P);
  col_stats<<<(K2P + 255) / 256, 256, 0, stream>>>(sig, K2P, mean, invs);
  fold_w<<<dim3((K2P + 255) / 256, GG), 256, 0, stream>>>(P[2][0], invs, S2, K2P, wf);
  fold_bias<<<GG, 256, 0, stream>>>(P[2][0], mean, invs, P[2][2], P[2][3], S2, biasf);
  gemm_wmma_split<<<dim3(MM / 16, NSPLIT), 32, 0, stream>>>(sig, wf, part, K2P);
  reduce_gates<<<(MM * GG + 255) / 256, 256, 0, stream>>>(part, biasf, gates);
  lstm2_scan<<<1, 512, 0, stream>>>(gates, P[2][1], P[2][4], P[2][5], P[2][6], P[2][7], o3);

  head_k<<<MM / 256, 256, 0, stream>>>(o3, lin_w, lin_b, (float*)d_out);
}
